// InputLayer_59450937311550
// MI455X (gfx1250) — compile-verified, matrix operand order checked by reading, not run on hardware
//
#include <hip/hip_runtime.h>

// ---------------------------------------------------------------------------
// GMM region log-prob for MI455X (gfx1250, wave32, WMMA).
//
// maha[r,k,b] = || L^-1 (x - mu) ||^2  ==  || M x - v ||^2,  M = L^-1, v = M mu
// => per region r: GEMM  [K*D=1024, D=32] x [D=32, B=2048]  (K-dim == 32
//    == exactly one v_wmma_f32_16x16x32_bf16 per 16x16 output tile),
//    with hi/lo bf16 operand splitting (3 wmma / tile) for ~fp32 accuracy.
// ---------------------------------------------------------------------------

typedef __bf16 bf16_t;
typedef __attribute__((ext_vector_type(16))) __bf16 v16bf;
typedef __attribute__((ext_vector_type(8)))  float  v8f;

#define RR   32
#define KK   32
#define DD   32
#define BB   2048
#define FF   1024
#define COLS 128   // batch columns per block
#define XPAD 48    // LDS row pad (bf16 elems): 96B stride keeps 32B alignment

static __device__ __forceinline__ float swap_half_wave(float x) {
  // ds_swizzle SWAPX16: xor_mask=0x10, and_mask=0x1f  (group-of-32 mode)
  return __int_as_float(
      __builtin_amdgcn_ds_swizzle(__float_as_int(x), 0x401F));
}

// ---------------------------------------------------------------------------
// Phase 1: per (r,k) build L (softplus diag), invert by forward substitution,
// emit bf16 hi/lo M (WMMA-A row-major layout), v = M*mu, logdet.
// One wave per (r,k); thread j solves column j of M (fully unrolled => regs).
// ---------------------------------------------------------------------------
__global__ __launch_bounds__(32)
void prep_kernel(const float* __restrict__ Lr, const float* __restrict__ mu,
                 bf16_t* __restrict__ Ah, bf16_t* __restrict__ Al,
                 float* __restrict__ Vv, float* __restrict__ Ld) {
  const int rk = blockIdx.x;        // r*K + k
  const int j  = threadIdx.x;       // 0..31

  __shared__ float Ls[DD][DD + 1];
  __shared__ float Ms[DD][DD + 1];
  __shared__ float dia[DD];
  __shared__ float mus[DD];

  const float* Lbase = Lr + (size_t)rk * DD * DD;
  float diagj = 0.f;
  #pragma unroll
  for (int t = 0; t < DD; ++t) {
    float v = Lbase[j * DD + t];
    if (t > j) v = 0.f;
    if (t == j) {
      float sp = (v > 20.f) ? v : log1pf(expf(v));   // softplus, stable
      diagj = sp + 1e-3f;
      v = diagj;
    }
    Ls[j][t] = v;
  }
  dia[j] = diagj;
  mus[j] = mu[(size_t)rk * DD + j];
  __syncthreads();

  // Solve L m = e_j (column j of M). Static indices -> pure registers.
  float m[DD];
  #pragma unroll
  for (int i = 0; i < DD; ++i) {
    float acc = (i == j) ? 1.0f : 0.0f;
    #pragma unroll
    for (int t = 0; t < DD; ++t) {
      if (t < i) acc -= Ls[i][t] * m[t];
    }
    m[i] = acc / Ls[i][i];
  }
  #pragma unroll
  for (int i = 0; i < DD; ++i) Ms[i][j] = m[i];
  __syncthreads();

  // Thread j now owns row j of M: contiguous 64B bf16 stores + v = M*mu.
  const size_t arow = ((size_t)rk * DD + j) * DD;
  float vj = 0.f;
  #pragma unroll
  for (int t = 0; t < DD; ++t) {
    const float val = Ms[j][t];
    const bf16_t h  = (bf16_t)val;
    Ah[arow + t] = h;
    Al[arow + t] = (bf16_t)(val - (float)h);
    vj += val * mus[t];
  }
  Vv[(size_t)rk * DD + j] = vj;
  if (j == 0) {
    float ld = 0.f;
    #pragma unroll
    for (int t = 0; t < DD; ++t) ld += logf(dia[t]);
    Ld[rk] = ld;
  }
}

// ---------------------------------------------------------------------------
// Phase 2: block = (region r, 128-column batch tile). Gather x into LDS with
// bf16 hi/lo split; 8 waves x 4 components; 3 wmma per 16x16 tile; squared-
// diff reduction; write lp[R,B,K].
// ---------------------------------------------------------------------------
__global__ __launch_bounds__(256)
void gmm_kernel(const float* __restrict__ inputs,
                const long long* __restrict__ regions,
                const bf16_t* __restrict__ Ah, const bf16_t* __restrict__ Al,
                const float* __restrict__ Vv, const float* __restrict__ Ld,
                float* __restrict__ out) {
  const int r  = blockIdx.x;
  const int cb = blockIdx.y * COLS;
  const int t  = threadIdx.x;

  __shared__ int cidx[DD];
  __shared__ alignas(32) bf16_t xh[COLS][XPAD];
  __shared__ alignas(32) bf16_t xl[COLS][XPAD];

  if (t < DD) cidx[t] = (int)regions[r * DD + t];
  __syncthreads();

  // Gather + split: x[col][d] = inputs[cb+col, regions[r][d]]
  for (int e = t; e < COLS * DD; e += 256) {
    const int col = e >> 5, d = e & 31;
    const float v = inputs[(size_t)(cb + col) * FF + cidx[d]];
    const bf16_t h = (bf16_t)v;
    xh[col][d] = h;
    xl[col][d] = (bf16_t)(v - (float)h);
  }
  __syncthreads();

  const int wave = t >> 5, lane = t & 31;
  const int lr = lane & 15, lh = lane >> 4;
  constexpr float LOG2PI = 1.8378770664093453f;

  for (int comp = 0; comp < 4; ++comp) {
    const int k    = wave * 4 + comp;
    const int rowg = (r * KK + k) * DD;   // first of 32 GEMM rows for (r,k)

    // WMMA A operands (16x32 bf16 = 8 VGPRs each): lane -> row rowg+lr,
    // k-elements [lh*16, lh*16+16). Rows are 64B, loads 32B-aligned.
    const v16bf a0h = *(const v16bf*)(Ah + ((size_t)(rowg + lr)) * DD + lh * 16);
    const v16bf a1h = *(const v16bf*)(Ah + ((size_t)(rowg + 16 + lr)) * DD + lh * 16);
    const v16bf a0l = *(const v16bf*)(Al + ((size_t)(rowg + lr)) * DD + lh * 16);
    const v16bf a1l = *(const v16bf*)(Al + ((size_t)(rowg + 16 + lr)) * DD + lh * 16);

    // v = M*mu broadcast per C-VGPR row (C layout: VGPR j -> row j / 8+j)
    float vr0[8], vr1[8];
    #pragma unroll
    for (int j = 0; j < 8; ++j) {
      vr0[j] = Vv[(size_t)(r * KK + k) * DD + lh * 8 + j];
      vr1[j] = Vv[(size_t)(r * KK + k) * DD + 16 + lh * 8 + j];
    }
    const float ld = Ld[r * KK + k];

    #pragma unroll
    for (int ct = 0; ct < COLS / 16; ++ct) {
      const int col = ct * 16 + lr;
      // B operand 32x16: lane -> column col, rows [lh*16, lh*16+16)
      const v16bf bh = *(const v16bf*)(&xh[col][lh * 16]);
      const v16bf bl = *(const v16bf*)(&xl[col][lh * 16]);

      v8f c0 = {};   // rows rowg..rowg+15
      c0 = __builtin_amdgcn_wmma_f32_16x16x32_bf16(false, a0h, false, bh, (short)0, c0, false, false);
      c0 = __builtin_amdgcn_wmma_f32_16x16x32_bf16(false, a0h, false, bl, (short)0, c0, false, false);
      c0 = __builtin_amdgcn_wmma_f32_16x16x32_bf16(false, a0l, false, bh, (short)0, c0, false, false);
      v8f c1 = {};   // rows rowg+16..rowg+31
      c1 = __builtin_amdgcn_wmma_f32_16x16x32_bf16(false, a1h, false, bh, (short)0, c1, false, false);
      c1 = __builtin_amdgcn_wmma_f32_16x16x32_bf16(false, a1h, false, bl, (short)0, c1, false, false);
      c1 = __builtin_amdgcn_wmma_f32_16x16x32_bf16(false, a1l, false, bh, (short)0, c1, false, false);

      // maha partial: this lane covers 16 of the 32 rows for its column.
      float s = 0.f;
      #pragma unroll
      for (int j = 0; j < 8; ++j) {
        const float d0 = c0[j] - vr0[j];
        const float d1 = c1[j] - vr1[j];
        s += d0 * d0 + d1 * d1;
      }
      s += swap_half_wave(s);   // combine lane L with lane L^16 -> all 32 rows

      if (lh == 0) {
        const float lp = -0.5f * ((float)DD * LOG2PI + s) - ld;
        out[((size_t)r * BB + cb + col) * KK + k] = lp;
      }
    }
  }
}

// ---------------------------------------------------------------------------
extern "C" void kernel_launch(void* const* d_in, const int* in_sizes, int n_in,
                              void* d_out, int out_size, void* d_ws, size_t ws_size,
                              hipStream_t stream) {
  (void)in_sizes; (void)n_in; (void)out_size; (void)ws_size;
  const float*     inputs  = (const float*)d_in[0];
  const float*     mu      = (const float*)d_in[1];
  const float*     L_raw   = (const float*)d_in[2];
  const long long* regions = (const long long*)d_in[3];
  float* out = (float*)d_out;

  // Workspace carve-out (~4.3 MB): Ah 2MB | Al 2MB | Vv 128KB | Ld 4KB
  char*   ws = (char*)d_ws;
  bf16_t* Ah = (bf16_t*)(ws);
  bf16_t* Al = (bf16_t*)(ws + (size_t)(2u << 20));
  float*  Vv = (float*) (ws + (size_t)(4u << 20));
  float*  Ld = (float*) (ws + (size_t)(4u << 20) + (size_t)(256u << 10));

  prep_kernel<<<dim3(RR * KK), dim3(32), 0, stream>>>(L_raw, mu, Ah, Al, Vv, Ld);
  gmm_kernel<<<dim3(RR, BB / COLS), dim3(256), 0, stream>>>(
      inputs, regions, Ah, Al, Vv, Ld, out);
}
